// SequenceCompressorAttention_21706764714099
// MI455X (gfx1250) — matrix operand order repaired and checked by reading, not compiled, not run on hardware
//
#include <hip/hip_runtime.h>

// ---------------------------------------------------------------------------
// SequenceCompressorAttention for MI455X (gfx1250, wave32, WMMA).
// Pipeline: LN -> [q,kv] bf16 WMMA GEMMs -> flash attention (WMMA) -> out GEMM.
// Matmuls: v_wmma_f32_16x16x32_bf16 (bf16 in, f32 accum).
// Data movement: global_load_async_to_lds_b128 (ASYNCcnt) with double
// buffering in the GEMM K-loop; s_wait_asynccnt fences before barriers.
// ---------------------------------------------------------------------------

typedef __bf16 bf16;
typedef __attribute__((ext_vector_type(16))) __bf16 bf16x16;
typedef __attribute__((ext_vector_type(8)))  __bf16 bf16x8;
typedef __attribute__((ext_vector_type(8)))  float  f32x8;

#define B_   8
#define N1_  4096
#define N2_  256
#define D_   1024
#define H_   16
#define DH_  64
#define NK_  4352   // N1 + N2

__device__ __forceinline__ f32x8 v8f_zero() {
  f32x8 z;
#pragma unroll
  for (int i = 0; i < 8; ++i) z[i] = 0.0f;
  return z;
}

__device__ __forceinline__ f32x8 wmma_bf16(bf16x16 a, bf16x16 b, f32x8 c) {
  return __builtin_amdgcn_wmma_f32_16x16x32_bf16(
      /*neg_a=*/false, a, /*neg_b=*/false, b,
      /*c_mod=*/(short)0, c, /*reuse_a=*/false, /*reuse_b=*/false);
}

// Async DMA: 16B global -> LDS, tracked by ASYNCcnt (no VGPR data path).
// VDST operand carries the LDS byte address (low 32 bits of flat shared ptr).
__device__ __forceinline__ void async_copy_b128(void* lds_ptr, const void* gptr) {
  unsigned int lds_off = (unsigned int)(unsigned long long)lds_ptr;
  asm volatile("global_load_async_to_lds_b128 %0, %1, off"
               :: "v"(lds_off), "v"(gptr)
               : "memory");
}
__device__ __forceinline__ void wait_async() {
  asm volatile("s_wait_asynccnt 0x0" ::: "memory");
}

// A operand (16xK, 16-bit): lane m = lane&15, group g = lane>>4.
// elems 0..7  -> K = 8g + 0..7 ; elems 8..15 -> K = 16 + 8g + 0..7
__device__ __forceinline__ bf16x16 frag_a(const bf16* base, int ldk) {
  const int lane = (int)(threadIdx.x & 31u);
  const int m = lane & 15, g = lane >> 4;
  const bf16* p = base + m * ldk + 8 * g;
  bf16x8 lo = *(const bf16x8*)(p);
  bf16x8 hi = *(const bf16x8*)(p + 16);
  bf16x16 f;
#pragma unroll
  for (int i = 0; i < 8; ++i) { f[i] = lo[i]; f[i + 8] = hi[i]; }
  return f;
}

// B operand (Kx16, 16-bit), stored [n][k] k-contiguous: lane n = lane&15,
// elems 0..7 -> K = 8g + 0..7 ; elems 8..15 -> K = 16 + 8g + 0..7
__device__ __forceinline__ bf16x16 frag_b(const bf16* base, int ldk) {
  const int lane = (int)(threadIdx.x & 31u);
  const int n = lane & 15, g = lane >> 4;
  const bf16* p = base + n * ldk + 8 * g;
  bf16x8 lo = *(const bf16x8*)(p);
  bf16x8 hi = *(const bf16x8*)(p + 16);
  bf16x16 f;
#pragma unroll
  for (int i = 0; i < 8; ++i) { f[i] = lo[i]; f[i + 8] = hi[i]; }
  return f;
}

// ---------------------------------------------------------------------------
// Kernel 1: W[K][N] f32  ->  Wt[N][K] bf16   (tiled transpose via LDS)
// ---------------------------------------------------------------------------
__global__ __launch_bounds__(256) void transpose_convert_kernel(
    const float* __restrict__ W, bf16* __restrict__ Wt, int K, int N) {
  __shared__ float t[32][33];
  const int n0 = blockIdx.x * 32, k0 = blockIdx.y * 32;
  const int tx = threadIdx.x & 31, ty = threadIdx.x >> 5;  // 32 x 8
#pragma unroll
  for (int r = ty; r < 32; r += 8)
    t[r][tx] = W[(long long)(k0 + r) * N + (n0 + tx)];
  __syncthreads();
#pragma unroll
  for (int r = ty; r < 32; r += 8)
    Wt[(long long)(n0 + r) * K + (k0 + tx)] = (bf16)t[tx][r];
}

// ---------------------------------------------------------------------------
// Kernel 2: row LayerNorm (D=1024), f32 in -> bf16 out into concat buffer
// ---------------------------------------------------------------------------
__global__ __launch_bounds__(256) void layernorm_kernel(
    const float* __restrict__ in, const float* __restrict__ gamma,
    const float* __restrict__ beta, bf16* __restrict__ dst,
    int rows_per_batch, int dst_row_off) {
  const int rid = blockIdx.x;
  const int b = rid / rows_per_batch;
  const int r = rid - b * rows_per_batch;
  const float* row = in + (long long)rid * D_;
  bf16* drow = dst + ((long long)b * NK_ + dst_row_off + r) * D_;

  const int tid = threadIdx.x;
  float4 xv = ((const float4*)row)[tid];
  float s = xv.x + xv.y + xv.z + xv.w;
  float ss = xv.x * xv.x + xv.y * xv.y + xv.z * xv.z + xv.w * xv.w;
#pragma unroll
  for (int off = 16; off > 0; off >>= 1) {
    s += __shfl_xor(s, off);
    ss += __shfl_xor(ss, off);
  }
  __shared__ float red[16];
  if ((tid & 31) == 0) { red[tid >> 5] = s; red[8 + (tid >> 5)] = ss; }
  __syncthreads();
  float S = 0.f, SS = 0.f;
#pragma unroll
  for (int w = 0; w < 8; ++w) { S += red[w]; SS += red[8 + w]; }
  const float mu = S * (1.0f / D_);
  const float var = SS * (1.0f / D_) - mu * mu;
  const float rstd = rsqrtf(var + 1e-5f);

  const int c = tid * 4;
  float4 gv = ((const float4*)gamma)[tid];
  float4 bv = ((const float4*)beta)[tid];
  drow[c + 0] = (bf16)((xv.x - mu) * rstd * gv.x + bv.x);
  drow[c + 1] = (bf16)((xv.y - mu) * rstd * gv.y + bv.y);
  drow[c + 2] = (bf16)((xv.z - mu) * rstd * gv.z + bv.z);
  drow[c + 3] = (bf16)((xv.w - mu) * rstd * gv.w + bv.w);
}

// ---------------------------------------------------------------------------
// Kernel 3: C[M][N] = A[M][K](bf16) x Wt[N][K](bf16), f32 accum, WMMA.
// Block tile 128x128, 8 waves, each wave 32x64. K step = 32.
// Double-buffered async (ASYNCcnt) global->LDS pipeline.
// ---------------------------------------------------------------------------
template <typename OutT>
__global__ __launch_bounds__(256) void gemm_bf16_kernel(
    const bf16* __restrict__ A, long long sA,
    const bf16* __restrict__ Wt,
    OutT* __restrict__ C, long long sC,
    int K, int ldc, float scale) {
  __shared__ bf16 As[2][128 * 32];
  __shared__ bf16 Bs[2][128 * 32];
  const int tid = threadIdx.x;
  const int lane = tid & 31, w = tid >> 5;
  const int wm = w & 3, wn = w >> 2;

  const bf16* Ab = A + (long long)blockIdx.z * sA + (long long)blockIdx.x * 128 * K;
  const bf16* Bb = Wt + (long long)blockIdx.y * 128 * K;
  OutT* Cb = C + (long long)blockIdx.z * sC;

  f32x8 acc[2][4];
#pragma unroll
  for (int mi = 0; mi < 2; ++mi)
#pragma unroll
    for (int nj = 0; nj < 4; ++nj) acc[mi][nj] = v8f_zero();

  // issue async copies of one 128x32 K-tile for A and B into buffer `buf`
  auto issue_tile = [&](int k0, int buf) {
#pragma unroll
    for (int c = tid; c < 512; c += 256) {
      const int r = c >> 2, c8 = (c & 3) << 3;
      async_copy_b128(&As[buf][r * 32 + c8], &Ab[(long long)r * K + k0 + c8]);
      async_copy_b128(&Bs[buf][r * 32 + c8], &Bb[(long long)r * K + k0 + c8]);
    }
  };

  issue_tile(0, 0);
  wait_async();
  __syncthreads();

  for (int k0 = 0; k0 < K; k0 += 32) {
    const int buf = (k0 >> 5) & 1;
    if (k0 + 32 < K) issue_tile(k0 + 32, buf ^ 1);  // overlap DMA with WMMA

    bf16x16 af0 = frag_a(&As[buf][(wm * 32 + 0) * 32], 32);
    bf16x16 af1 = frag_a(&As[buf][(wm * 32 + 16) * 32], 32);
#pragma unroll
    for (int nj = 0; nj < 4; ++nj) {
      bf16x16 bf = frag_b(&Bs[buf][(wn * 64 + nj * 16) * 32], 32);
      acc[0][nj] = wmma_bf16(af0, bf, acc[0][nj]);
      acc[1][nj] = wmma_bf16(af1, bf, acc[1][nj]);
    }

    wait_async();     // next tile's DMA complete (this wave)
    __syncthreads();  // visible to all waves; prev buffer free for reuse
  }

  // C layout: lane n = lane&15 is column; vgpr i holds row i + 8*(lane>=16).
  const int n = lane & 15, g = lane >> 4;
#pragma unroll
  for (int mi = 0; mi < 2; ++mi)
#pragma unroll
    for (int nj = 0; nj < 4; ++nj) {
      const int col = blockIdx.y * 128 + wn * 64 + nj * 16 + n;
#pragma unroll
      for (int i = 0; i < 8; ++i) {
        const long long row = (long long)blockIdx.x * 128 + wm * 32 + mi * 16 + i + 8 * g;
        Cb[row * ldc + col] = (OutT)(acc[mi][nj][i] * scale);
      }
    }
}

// ---------------------------------------------------------------------------
// Kernel 4: flash attention. Block = 128 q rows of one (b,h).
// q already pre-scaled by DH^-0.5. kv[b][j][0:1024]=K, [1024:2048]=V.
// Q/K tiles staged via async DMA; V transposed manually; mask staged to LDS.
// ---------------------------------------------------------------------------
#define VT_LD 136  // padded V^T row stride (elems), 16B-aligned rows

__global__ __launch_bounds__(256) void attention_kernel(
    const bf16* __restrict__ q,    // [B][256][1024]
    const bf16* __restrict__ kv,   // [B][4352][2048]
    const int* __restrict__ mask,  // [B][4352]
    bf16* __restrict__ out) {      // [B][256][1024]  (b, i, h*64+d)
  extern __shared__ __align__(128) char smem_raw[];
  bf16* Qs  = (bf16*)smem_raw;          // [128][64]
  bf16* Ks  = Qs + 128 * 64;            // [128][64]   key-major (B-op for QK^T)
  bf16* Vts = Ks + 128 * 64;            // [64][VT_LD] dh-major  (B-op for PV)
  bf16* Ps  = Vts + 64 * VT_LD;         // [8 waves][16][128] (A-op for PV)
  float* Ms = (float*)(Ps + 8 * 16 * 128);  // [128] mask penalties

  const int qt = blockIdx.x, h = blockIdx.y, b = blockIdx.z;
  const int tid = threadIdx.x;
  const int lane = tid & 31, w = tid >> 5;
  const int n = lane & 15, g = lane >> 4;

  // --- async-load Q tile (128 x 64) once; completion covered by first fence
  const bf16* qbase = q + ((long long)b * N2_ + qt * 128) * (H_ * DH_) + h * DH_;
#pragma unroll
  for (int c = tid; c < 1024; c += 256) {
    const int row = c >> 3, c8 = (c & 7) << 3;
    async_copy_b128(&Qs[row * 64 + c8], &qbase[(long long)row * (H_ * DH_) + c8]);
  }

  float m_i[8], l_i[8];
  f32x8 o[4];
#pragma unroll
  for (int i = 0; i < 8; ++i) { m_i[i] = -1e30f; l_i[i] = 0.0f; }
#pragma unroll
  for (int d = 0; d < 4; ++d) o[d] = v8f_zero();

  bf16* pb = Ps + w * 16 * 128;
  const bf16* qa = Qs + (w * 16) * 64;

  for (int kt = 0; kt < NK_ / 128; ++kt) {
    __syncthreads();  // previous tile fully consumed by all waves
    // --- K tile (128 keys x 64 dh), key-major, async DMA
    const bf16* kbase = kv + ((long long)b * NK_ + kt * 128) * (2 * H_ * DH_) + h * DH_;
#pragma unroll
    for (int c = tid; c < 1024; c += 256) {
      const int row = c >> 3, c8 = (c & 7) << 3;
      async_copy_b128(&Ks[row * 64 + c8], &kbase[(long long)row * (2 * H_ * DH_) + c8]);
    }
    // --- V tile transposed into [dh][key] (synchronous: needs transpose)
    const bf16* vbase = kbase + H_ * DH_;
#pragma unroll
    for (int c = tid; c < 1024; c += 256) {
      const int key = c >> 3, d8 = (c & 7) << 3;
      bf16x8 vv = *(const bf16x8*)&vbase[(long long)key * (2 * H_ * DH_) + d8];
#pragma unroll
      for (int e = 0; e < 8; ++e) Vts[(d8 + e) * VT_LD + key] = vv[e];
    }
    // --- mask penalties for this tile
    if (tid < 128)
      Ms[tid] = (mask[(long long)b * NK_ + kt * 128 + tid] == 0) ? -10000.0f : 0.0f;
    if (kt + 1 < NK_ / 128)  // hint next V tile toward the caches
      __builtin_prefetch(&vbase[(long long)(128 + (tid >> 1)) * (2 * H_ * DH_)], 0, 1);

    wait_async();
    __syncthreads();

    // --- S = Q (16x64) x K^T (64x128): 8 col-tiles, 2 K-steps each
    f32x8 s[8];
#pragma unroll
    for (int jt = 0; jt < 8; ++jt) s[jt] = v8f_zero();
#pragma unroll
    for (int kk = 0; kk < 2; ++kk) {
      bf16x16 a = frag_a(qa + kk * 32, 64);
#pragma unroll
      for (int jt = 0; jt < 8; ++jt) {
        bf16x16 bf = frag_b(Ks + (jt * 16) * 64 + kk * 32, 64);
        s[jt] = wmma_bf16(a, bf, s[jt]);
      }
    }

    // --- mask penalty (reference: masked positions get -10000 pre-softmax)
#pragma unroll
    for (int jt = 0; jt < 8; ++jt) {
      const float pen = Ms[jt * 16 + n];
#pragma unroll
      for (int i = 0; i < 8; ++i) s[jt][i] += pen;
    }

    // --- online softmax. Row i+8g lives in vgpr i across 16 lanes of a half.
    float mnew[8], corr[8], acc[8];
#pragma unroll
    for (int i = 0; i < 8; ++i) {
      float t = s[0][i];
#pragma unroll
      for (int jt = 1; jt < 8; ++jt) t = fmaxf(t, s[jt][i]);
#pragma unroll
      for (int off = 8; off > 0; off >>= 1) t = fmaxf(t, __shfl_xor(t, off));
      mnew[i] = fmaxf(m_i[i], t);
      corr[i] = __expf(m_i[i] - mnew[i]);
      acc[i] = 0.0f;
    }
#pragma unroll
    for (int d = 0; d < 4; ++d)
#pragma unroll
      for (int i = 0; i < 8; ++i) o[d][i] *= corr[i];
#pragma unroll
    for (int jt = 0; jt < 8; ++jt)
#pragma unroll
      for (int i = 0; i < 8; ++i) {
        const float p = __expf(s[jt][i] - mnew[i]);
        s[jt][i] = p;
        acc[i] += p;
      }
#pragma unroll
    for (int i = 0; i < 8; ++i) {
      float t = acc[i];
#pragma unroll
      for (int off = 8; off > 0; off >>= 1) t += __shfl_xor(t, off);
      l_i[i] = l_i[i] * corr[i] + t;
      m_i[i] = mnew[i];
    }

    // --- stage P (16x128) to per-wave LDS in A-operand layout
#pragma unroll
    for (int jt = 0; jt < 8; ++jt)
#pragma unroll
      for (int i = 0; i < 8; ++i)
        pb[(i + 8 * g) * 128 + jt * 16 + n] = (bf16)s[jt][i];

    // --- O += P (16x128) x V (128x64): 4 K-steps, 4 dh tiles
#pragma unroll
    for (int kk = 0; kk < 4; ++kk) {
      bf16x16 a = frag_a(pb + kk * 32, 128);
#pragma unroll
      for (int d = 0; d < 4; ++d) {
        bf16x16 bf = frag_b(Vts + (d * 16) * VT_LD + kk * 32, VT_LD);
        o[d] = wmma_bf16(a, bf, o[d]);
      }
    }
  }

  // --- normalize and write [b][row][h*64+dh]
  float linv[8];
#pragma unroll
  for (int i = 0; i < 8; ++i) linv[i] = 1.0f / l_i[i];
  bf16* ob = out + ((long long)b * N2_ + qt * 128 + w * 16) * (H_ * DH_) + h * DH_;
#pragma unroll
  for (int d = 0; d < 4; ++d)
#pragma unroll
    for (int i = 0; i < 8; ++i) {
      const int r = i + 8 * g;
      ob[(long long)r * (H_ * DH_) + d * 16 + n] = (bf16)(o[d][i] * linv[i]);
    }
}

// ---------------------------------------------------------------------------
// Host-side launcher
// ---------------------------------------------------------------------------
extern "C" void kernel_launch(void* const* d_in, const int* in_sizes, int n_in,
                              void* d_out, int out_size, void* d_ws, size_t ws_size,
                              hipStream_t stream) {
  (void)in_sizes; (void)n_in; (void)out_size; (void)ws_size;
  const float* x       = (const float*)d_in[0];
  const float* latents = (const float*)d_in[1];
  const int*   mask    = (const int*)d_in[2];
  const float* g_media = (const float*)d_in[3];
  const float* b_media = (const float*)d_in[4];
  const float* g_lat   = (const float*)d_in[5];
  const float* b_lat   = (const float*)d_in[6];
  const float* Wq      = (const float*)d_in[7];
  const float* Wkv     = (const float*)d_in[8];
  const float* Wout    = (const float*)d_in[9];
  float* out = (float*)d_out;

  // workspace carve (bf16 buffers), 256B aligned
  char* ws = (char*)d_ws;
  size_t off = 0;
  auto carve = [&](size_t elems) {
    bf16* p = (bf16*)(ws + off);
    off = (off + elems * sizeof(bf16) + 255) & ~(size_t)255;
    return p;
  };
  bf16* src   = carve((size_t)B_ * NK_ * D_);        // LN(x) ++ LN(latents)
  bf16* qb    = carve((size_t)B_ * N2_ * D_);        // q (pre-scaled)
  bf16* kvb   = carve((size_t)B_ * NK_ * 2 * D_);    // k|v
  bf16* aob   = carve((size_t)B_ * N2_ * D_);        // attention output
  bf16* wqt   = carve((size_t)D_ * D_);              // Wq^T
  bf16* wkvt  = carve((size_t)2 * D_ * D_);          // Wkv^T
  bf16* woutt = carve((size_t)D_ * D_);              // Wout^T

  // 1) weights -> bf16, transposed to [N][K]
  transpose_convert_kernel<<<dim3(D_ / 32, D_ / 32), 256, 0, stream>>>(Wq, wqt, D_, D_);
  transpose_convert_kernel<<<dim3(2 * D_ / 32, D_ / 32), 256, 0, stream>>>(Wkv, wkvt, D_, 2 * D_);
  transpose_convert_kernel<<<dim3(D_ / 32, D_ / 32), 256, 0, stream>>>(Wout, woutt, D_, D_);

  // 2) LayerNorms into concat buffer
  layernorm_kernel<<<B_ * N1_, 256, 0, stream>>>(x, g_media, b_media, src, N1_, 0);
  layernorm_kernel<<<B_ * N2_, 256, 0, stream>>>(latents, g_lat, b_lat, src, N2_, N1_);

  // 3) q = ln @ Wq * DH^-0.5   (M=256/batch)
  gemm_bf16_kernel<bf16><<<dim3(N2_ / 128, D_ / 128, B_), 256, 0, stream>>>(
      src + (long long)N1_ * D_, (long long)NK_ * D_, wqt,
      qb, (long long)N2_ * D_, D_, D_, 0.125f);

  // 4) kv = concat @ Wkv       (M=4352/batch, N=2048)
  gemm_bf16_kernel<bf16><<<dim3(NK_ / 128, 2 * D_ / 128, B_), 256, 0, stream>>>(
      src, (long long)NK_ * D_, wkvt,
      kvb, (long long)NK_ * 2 * D_, D_, 2 * D_, 1.0f);

  // 5) flash attention
  const size_t smem = (size_t)(128 * 64 + 128 * 64 + 64 * VT_LD + 8 * 16 * 128) * sizeof(bf16)
                    + 128 * sizeof(float);
  attention_kernel<<<dim3(N2_ / 128, H_, B_), 256, smem, stream>>>(qb, kvb, mask, aob);

  // 6) out = attn_out @ Wout  (flattened M = B*N2 = 2048, f32 -> d_out)
  gemm_bf16_kernel<float><<<dim3(B_ * N2_ / 128, D_ / 128, 1), 256, 0, stream>>>(
      aob, 0, woutt, out, 0, D_, D_, 1.0f);
}
